// VectorQuantizer_22806276341897
// MI455X (gfx1250) — matrix operand order repaired and checked by reading, not compiled, not run on hardware
//
#include <hip/hip_runtime.h>

// ---------------------------------------------------------------------------
// VectorQuantizer fused kernel set for MI455X (gfx1250, wave32, WMMA)
//   x:        [16, 512, 64, 16] fp32    (BS, C, F, J)
//   codebook: [8192, 512]       fp32
//   out:      [16, 512, 64, 16] fp32  ++  commit_loss scalar  (8388609 floats)
// ---------------------------------------------------------------------------

typedef _Float16 v16h __attribute__((ext_vector_type(16)));
typedef _Float16 v8h  __attribute__((ext_vector_type(8)));
typedef float    v8f  __attribute__((ext_vector_type(8)));

#define NB_CODE   8192
#define CODE_DIM  512
#define N_ROWS    16384        // BS*F*J = 16*64*16
#define BROWS     32           // rows per block in the argmin GEMM
#define APITCH    520          // LDS row pitch in f16 (512 + 8 pad -> conflict-free)

// float -> order-preserving unsigned key (ascending float == ascending uint)
__device__ __forceinline__ unsigned sortkey(float s) {
    unsigned u = __float_as_uint(s);
    return (u & 0x80000000u) ? ~u : (u | 0x80000000u);
}

// --- kernel 0: zero the loss accumulator ----------------------------------
__global__ void vq_zero(float* p) {
    if (threadIdx.x == 0 && blockIdx.x == 0) *p = 0.0f;
}

// --- kernel 1: codebook fp32 -> f16, plus ||k||^2 -------------------------
__global__ void vq_prep_codebook(const float* __restrict__ cb,
                                 _Float16* __restrict__ cb16,
                                 float* __restrict__ cbnorm) {
    __shared__ float red[256];
    const int code = blockIdx.x;
    const int t = threadIdx.x;
    const float*     src = cb   + (size_t)code * CODE_DIM;
    _Float16*        dst = cb16 + (size_t)code * CODE_DIM;
    float s = 0.0f;
#pragma unroll
    for (int i = 0; i < 2; ++i) {
        float v = src[t + 256 * i];
        dst[t + 256 * i] = (_Float16)v;
        s += v * v;
    }
    red[t] = s;
    __syncthreads();
    for (int off = 128; off > 0; off >>= 1) {
        if (t < off) red[t] += red[t + off];
        __syncthreads();
    }
    if (t == 0) cbnorm[code] = red[0];
}

// --- kernel 2: x [b,c,f,j] fp32 -> xf16 [n, c] f16 ------------------------
// n = b*1024 + f*16 + j ;  x flat index = (b*512 + c)*1024 + (f*16 + j)
__global__ void vq_prep_x(const float* __restrict__ x,
                          _Float16* __restrict__ xf16) {
    const int g = blockIdx.x * blockDim.x + threadIdx.x;   // over N_ROWS*CODE_DIM
    const int n  = g >> 9;          // /512
    const int c  = g & 511;
    const int b  = n >> 10;         // /1024
    const int fj = n & 1023;        // f*16 + j
    const size_t src = ((size_t)(b * 512 + c) << 10) + (size_t)fj;
    xf16[g] = (_Float16)x[src];
}

// --- kernel 3: fused f16 WMMA GEMM + running argmin -----------------------
// Block: 512 threads = 16 waves. 2 row-tiles (16 rows each) x 8 code streams.
// score[n, k] = ||k||^2 - 2 * dot(xf[n], cb[k])   (||x||^2 is row-constant)
__global__ void __launch_bounds__(512, 1)
vq_argmin(const _Float16* __restrict__ xf16,
          const _Float16* __restrict__ cb16,
          const float* __restrict__ cbnorm,
          unsigned* __restrict__ idx) {
    __shared__ __align__(16) _Float16 Alds[BROWS * APITCH];
    __shared__ unsigned long long best[BROWS];

    const int t = threadIdx.x;
    if (t < BROWS) best[t] = ~0ull;

    // stage A tile (32 rows x 512 f16) into LDS: thread t -> row t/16, 32-col chunk
    {
        const int r  = t >> 4;              // 0..31
        const int c0 = (t & 15) * 32;
        const _Float16* src = xf16 + ((size_t)blockIdx.x * BROWS + r) * CODE_DIM + c0;
        _Float16*       dst = Alds + r * APITCH + c0;
#pragma unroll
        for (int i = 0; i < 4; ++i)
            *(v8h*)(dst + i * 8) = *(const v8h*)(src + i * 8);
    }
    __syncthreads();

    const int lane = t & 31;
    const int w    = t >> 5;      // wave id 0..15
    const int rt   = w & 1;       // row tile 0..1
    const int cs   = w >> 1;      // code stream 0..7
    const int m    = lane & 15;   // row / column within tile
    const int h    = lane >> 4;   // lane half

    // A fragment per lane-half: K = [8h, 8h+8) U [16+8h, 16+8h+8)  (ISA 16-bit A layout)
    const _Float16* arow = Alds + (rt * 16 + m) * APITCH + 8 * h;

    unsigned long long bl[8];
#pragma unroll
    for (int r = 0; r < 8; ++r) bl[r] = ~0ull;

    for (int ct = cs; ct < NB_CODE / 16; ct += 8) {
        const int code = ct * 16 + m;
        // B fragment per lane-half: K = [8h, 8h+8) U [16+8h, 16+8h+8) of codebook row `code`
        const _Float16* brow = cb16 + (size_t)code * CODE_DIM + 8 * h;
        // prefetch this stream's code tile 16 steps (2048 codes) ahead (gfx1250 global_prefetch_b8)
        __builtin_prefetch(brow + (size_t)128 * 16 * CODE_DIM, 0, 0);

        v8f acc = {};
#pragma unroll
        for (int ks = 0; ks < 16; ++ks) {
            const int kb = ks * 32;
            v8h  al = *(const v8h*)(arow + kb);
            v8h  ah = *(const v8h*)(arow + kb + 16);
            v16h a  = __builtin_shufflevector(al, ah,
                        0, 1, 2, 3, 4, 5, 6, 7, 8, 9, 10, 11, 12, 13, 14, 15);
            v8h  bl8 = *(const v8h*)(brow + kb);
            v8h  bh8 = *(const v8h*)(brow + kb + 16);
            v16h b  = __builtin_shufflevector(bl8, bh8,
                        0, 1, 2, 3, 4, 5, 6, 7, 8, 9, 10, 11, 12, 13, 14, 15);
            // D = A(16x32,f16) * B(32x16,f16) + C(f32)
            acc = __builtin_amdgcn_wmma_f32_16x16x32_f16(
                      false, a, false, b, (short)0, acc, false, false);
        }

        const float cn = cbnorm[code];
#pragma unroll
        for (int r = 0; r < 8; ++r) {
            const float score = cn - 2.0f * acc[r];           // row m' = r + 8h, col = code
            const unsigned long long packed =
                ((unsigned long long)sortkey(score) << 32) | (unsigned)code;
            bl[r] = (packed < bl[r]) ? packed : bl[r];
        }
    }

    // merge lane-local bests across lanes/waves: min of packed u64 == argmin
#pragma unroll
    for (int r = 0; r < 8; ++r)
        atomicMin(&best[rt * 16 + r + 8 * h], bl[r]);
    __syncthreads();

    if (t < BROWS)
        idx[(size_t)blockIdx.x * BROWS + t] = (unsigned)(best[t] & 0xFFFFFFFFull);
}

// --- kernel 4: gather output (fp32) + commit-loss partial sums ------------
__global__ void vq_output(const float* __restrict__ x,
                          const float* __restrict__ cb,
                          const unsigned* __restrict__ idx,
                          float* __restrict__ out,
                          float* __restrict__ loss) {
    __shared__ float red[256];
    const size_t e = (size_t)blockIdx.x * 256 + threadIdx.x;  // flat [b,c,f,j]
    const int j = (int)(e & 15);
    const int f = (int)((e >> 4) & 63);
    const int c = (int)((e >> 10) & 511);
    const int b = (int)(e >> 19);
    const int n = (b << 10) + (f << 4) + j;
    const unsigned k = idx[n];
    const float xd = cb[(size_t)k * CODE_DIM + c];
    const float xv = x[e];
    out[e] = xd;                       // x_st == x_d numerically (straight-through)
    const float d = xv - xd;
    red[threadIdx.x] = d * d;
    __syncthreads();
    for (int off = 128; off > 0; off >>= 1) {
        if (threadIdx.x < off) red[threadIdx.x] += red[threadIdx.x + off];
        __syncthreads();
    }
    if (threadIdx.x == 0) atomicAdd(loss, red[0]);
}

// --- kernel 5: finalize loss ----------------------------------------------
__global__ void vq_final(const float* __restrict__ loss, float* __restrict__ out_loss) {
    *out_loss = *loss * (1.0f / (float)((size_t)N_ROWS * CODE_DIM));
}

// ---------------------------------------------------------------------------
extern "C" void kernel_launch(void* const* d_in, const int* in_sizes, int n_in,
                              void* d_out, int out_size, void* d_ws, size_t ws_size,
                              hipStream_t stream) {
    const float* x  = (const float*)d_in[0];   // [16,512,64,16]
    const float* cb = (const float*)d_in[1];   // [8192,512]

    char* ws = (char*)d_ws;
    _Float16* cb16   = (_Float16*)(ws);                         //  8,388,608 B
    _Float16* xf16   = (_Float16*)(ws + 8388608);               // 16,777,216 B
    float*    cbnorm = (float*)   (ws + 25165824);              //     32,768 B
    unsigned* idx    = (unsigned*)(ws + 25198592);              //     65,536 B
    float*    lossac = (float*)   (ws + 25264128);              //          4 B

    float* out      = (float*)d_out;
    float* out_loss = out + (size_t)N_ROWS * CODE_DIM;          // d_out[8388608]

    vq_zero<<<1, 1, 0, stream>>>(lossac);
    vq_prep_codebook<<<NB_CODE, 256, 0, stream>>>(cb, cb16, cbnorm);
    vq_prep_x<<<(N_ROWS * CODE_DIM) / 256, 256, 0, stream>>>(x, xf16);
    vq_argmin<<<N_ROWS / BROWS, 512, 0, stream>>>(xf16, cb16, cbnorm, idx);
    vq_output<<<(N_ROWS * CODE_DIM) / 256, 256, 0, stream>>>(x, cb, idx, out, lossac);
    vq_final<<<1, 1, 0, stream>>>(lossac, out_loss);
}
